// LinearSSMAttention_24979529794315
// MI455X (gfx1250) — matrix operand order repaired and checked by reading, not compiled, not run on hardware
//
#include <hip/hip_runtime.h>

// ---------------------------------------------------------------------------
// LinearSSM on MI455X (gfx1250, wave32, WMMA + async-LDS):
//   U = X @ B           bf16 WMMA GEMM (65536x448x448)
//   h_t = h_{t-1}@A+u_t single-WG recurrence; A slices live in VGPRs as
//                       fragment-major bf16 WMMA operands; u_t prefetched /
//                       h_t written back with async LDS<->global b128 ops;
//                       U overwritten in place by H (stays L2-resident)
//   Y = H @ C^T         bf16 WMMA GEMM (same kernel as U = X@B)
// Weights are pre-packed FRAGMENT-MAJOR: [kstep][ntile][lane][g] packed bf16
// pairs, so LDS staging is a verbatim async copy and each WMMA operand is
// exactly two ds_load_b128. All hot loops are software-pipelined (double-
// buffered LDS tiles + double-buffered operand fragments) so ds/async
// traffic overlaps the WMMA pipe instead of full-drain waits.
// ---------------------------------------------------------------------------

#define STATE 448
#define SEQ   4096
#define BATCH 16
#define MROWS (BATCH * SEQ)     // 65536
#define KP    (STATE / 2)       // 224 bf16 K-pairs
#define NT    (STATE / 16)      // 28 N tiles
#define KT    (STATE / 32)      // 14 K steps of 32
#define FRAG_TILE (NT * 256)    // u32 per k-step of fragment-major weights

typedef __attribute__((ext_vector_type(16))) __bf16 v16bf;
typedef __attribute__((ext_vector_type(8)))  float  v8f;
typedef __attribute__((ext_vector_type(4)))  int    v4i;

union Frag { unsigned u[8]; uint4 q[2]; v16bf b; };

// ------------------------- async LDS <-> global ----------------------------
#if defined(__has_builtin) && __has_builtin(__builtin_amdgcn_global_load_async_to_lds_b128)
#define HAVE_ASYNC_LOAD 1
#else
#define HAVE_ASYNC_LOAD 0
#endif
#if defined(__has_builtin) && __has_builtin(__builtin_amdgcn_global_store_async_from_lds_b128)
#define HAVE_ASYNC_STORE 1
#else
#define HAVE_ASYNC_STORE 0
#endif
#if defined(__has_builtin) && __has_builtin(__builtin_amdgcn_s_wait_asynccnt)
#define HAVE_WAIT_ASYNC 1
#else
#define HAVE_WAIT_ASYNC 0
#endif

// Builtin signature (probe-confirmed): param0 = v4i addrspace(1)* (global),
// param1 = v4i addrspace(3)* (LDS), then two immediate ints (offset, cpol).
typedef __attribute__((address_space(1))) v4i as1_v4i;
typedef __attribute__((address_space(3))) v4i as3_v4i;

__device__ __forceinline__ void async_g2l_b128(void* lds, const void* g) {
#if HAVE_ASYNC_LOAD
  __builtin_amdgcn_global_load_async_to_lds_b128(
      (as1_v4i*)g, (as3_v4i*)lds, 0, 0);
#else
  *(uint4*)lds = *(const uint4*)g;            // sync fallback
#endif
}
__device__ __forceinline__ void async_l2g_b128(void* g, const void* lds) {
#if HAVE_ASYNC_STORE
  __builtin_amdgcn_global_store_async_from_lds_b128(
      (as1_v4i*)g, (as3_v4i*)lds, 0, 0);
#else
  *(uint4*)g = *(const uint4*)lds;            // sync fallback
#endif
}
__device__ __forceinline__ void wait_async0() {
#if HAVE_ASYNC_LOAD || HAVE_ASYNC_STORE
#if HAVE_WAIT_ASYNC
  __builtin_amdgcn_s_wait_asynccnt(0);
#else
  asm volatile("s_wait_asynccnt 0x0" ::: "memory");
#endif
#endif
}

// ------------------------------- bf16 pack ---------------------------------
__device__ __forceinline__ unsigned short f2bf(float f) {
  union { float f; unsigned u; } v; v.f = f;
  unsigned r = v.u + 0x7FFFu + ((v.u >> 16) & 1u);   // round-to-nearest-even
  return (unsigned short)(r >> 16);
}
__device__ __forceinline__ unsigned pack2(float lo, float hi) {
  return (unsigned)f2bf(lo) | ((unsigned)f2bf(hi) << 16);
}

// ---------------------------------------------------------------------------
// Pre-pack A, B, C^T into FRAGMENT-MAJOR bf16 B-operand layout:
//   idx = ((ks*28 + nt)*32 + lane)*8 + g
//   kp  = 16*ks + g + 8*(lane>>4)          (contiguous k order: k = 16*(lane>>4) + s)
//   n   = nt*16 + (lane&15)
//   val = {bf16(W[2kp][n]), bf16(W[2kp+1][n])}     (C is packed transposed)
// ---------------------------------------------------------------------------
__global__ void prep_pack(const float* __restrict__ A,
                          const float* __restrict__ B,
                          const float* __restrict__ C,
                          unsigned* __restrict__ Afrag,
                          unsigned* __restrict__ Bfrag,
                          unsigned* __restrict__ Cfrag) {
  int i = blockIdx.x * blockDim.x + threadIdx.x;
  if (i >= KT * FRAG_TILE) return;
  int g = i & 7, l = (i >> 3) & 31, r = i >> 8;
  int nt = r % NT, ks = r / NT;
  int kp = 16 * ks + g + 8 * (l >> 4);
  int n  = nt * 16 + (l & 15);
  int k0 = 2 * kp, k1 = 2 * kp + 1;
  Afrag[i] = pack2(A[(size_t)k0 * STATE + n], A[(size_t)k1 * STATE + n]);
  Bfrag[i] = pack2(B[(size_t)k0 * STATE + n], B[(size_t)k1 * STATE + n]);
  Cfrag[i] = pack2(C[(size_t)n * STATE + k0], C[(size_t)n * STATE + k1]);
}

// ---------------------------------------------------------------------------
// out[M x 448] = in[M x 448] @ W, W given fragment-major packed bf16.
// 256 threads = 8 waves; each wave: 16 M rows x all 28 N tiles (28 x v8f).
// Double-buffered LDS tiles: async copy of W tile ks+1 and f32->bf16 staging
// of X tile ks+1 overlap the 28-WMMA compute of tile ks (1 barrier/k-step).
// ---------------------------------------------------------------------------
__global__ __launch_bounds__(256) void gemm_bf16_448(
    const float* __restrict__ in,        // M x 448 f32
    const unsigned* __restrict__ wfrag,  // 14 x 7168 fragment-major u32
    float* __restrict__ out) {           // M x 448 f32
  __shared__ unsigned bTf[2][FRAG_TILE]; // 2 x 28,672 B verbatim W tiles
  __shared__ unsigned xT[2][128][20];    // 2 x (m-local x kp-local, padded)

  const int tid  = threadIdx.x;
  const int wave = tid >> 5;
  const int lane = tid & 31;
  const int ln   = lane & 15;
  const int side = lane >> 4;
  const int m0   = blockIdx.x * 128;

  v8f acc[NT];
#pragma unroll
  for (int nt = 0; nt < NT; ++nt)
#pragma unroll
    for (int r = 0; r < 8; ++r) acc[nt][r] = 0.0f;

  // Prologue: stage tiles for ks = 0.
  for (int i = tid; i < FRAG_TILE / 4; i += 256)
    async_g2l_b128(&bTf[0][i * 4], wfrag + i * 4);
  for (int i = tid; i < 128 * 16; i += 256) {
    int m = i >> 4, kp = i & 15;
    const float* xp = in + (size_t)(m0 + m) * STATE + 2 * kp;
    xT[0][m][kp] = pack2(xp[0], xp[1]);
  }

  for (int ks = 0; ks < KT; ++ks) {
    const int buf = ks & 1;
    wait_async0();
    __syncthreads();                       // tiles[ks] ready for all waves

    if (ks + 1 < KT) {                     // stage tiles[ks+1] (other buffer)
      const unsigned* src = wfrag + (size_t)(ks + 1) * FRAG_TILE;
      for (int i = tid; i < FRAG_TILE / 4; i += 256)
        async_g2l_b128(&bTf[buf ^ 1][i * 4], src + i * 4);
      for (int i = tid; i < 128 * 16; i += 256) {
        int m = i >> 4, kp = i & 15;
        const float* xp =
            in + (size_t)(m0 + m) * STATE + 32 * (ks + 1) + 2 * kp;
        xT[buf ^ 1][m][kp] = pack2(xp[0], xp[1]);
      }
    }

    // A-operand (16x32 bf16): K-pairs {8s..8s+7} -> 2x b128.
    Frag a;
    {
      const unsigned* row = &xT[buf][wave * 16 + ln][0];
      a.q[0] = *(const uint4*)(row + 8 * side);
      a.q[1] = *(const uint4*)(row + 8 * side + 4);
    }
    // Software-pipelined N-tile loop: load fragment nt+1 while WMMA nt runs.
    Frag w[2];
    {
      const unsigned* bp = &bTf[buf][lane * 8];
      w[0].q[0] = *(const uint4*)(bp);
      w[0].q[1] = *(const uint4*)(bp + 4);
    }
#pragma unroll
    for (int nt = 0; nt < NT; ++nt) {
      if (nt + 1 < NT) {
        const unsigned* bp = &bTf[buf][((nt + 1) * 32 + lane) * 8];
        w[(nt + 1) & 1].q[0] = *(const uint4*)(bp);
        w[(nt + 1) & 1].q[1] = *(const uint4*)(bp + 4);
      }
      acc[nt] = __builtin_amdgcn_wmma_f32_16x16x32_bf16(
          false, a.b, false, w[nt & 1].b, (short)0, acc[nt], false, false);
    }
  }

  // C/D layout: lane -> N = lane&15; VGPR r -> M = r + 8*side.
  const int mrow = m0 + wave * 16 + 8 * side;
#pragma unroll
  for (int nt = 0; nt < NT; ++nt) {
    float* op = out + (size_t)mrow * STATE + nt * 16 + ln;
#pragma unroll
    for (int r = 0; r < 8; ++r) op[(size_t)r * STATE] = acc[nt][r];
  }
}

// ---------------------------------------------------------------------------
// Recurrence: single WG, 28 waves = 28 n-tiles covering all 448 columns.
// Per wave: A[:,16w:16w+16] preloaded as 14 bf16 fragments (112 VGPRs).
// Per step: u_t comes from an async-prefetched LDS double buffer; 14 chained
// v_wmma_f32_16x16x32_bf16 (h-fragment loads software-pipelined) add
// h_{t-1}@A; h_t goes to an f32 LDS mirror, is re-packed to the bf16 operand
// buffer for t+1, and async-stored back over u_t in global (U becomes H).
// ---------------------------------------------------------------------------
__global__ __launch_bounds__(896) void ssm_recurrence(
    float* __restrict__ U,                  // (16,4096,448) f32, in-place
    const unsigned* __restrict__ Afrag) {   // fragment-major packed A
  __shared__ unsigned hb2[2][16][228];      // m x kp (padded), bf16 pairs
  __shared__ float    hF[16][452];          // m x state f32 mirror of h_t
  __shared__ float    uB[2][16][452];       // async u_t staging buffers

  const int tid  = threadIdx.x;
  const int wave = tid >> 5;     // 0..27 -> n-tile
  const int lane = tid & 31;
  const int ln   = lane & 15;
  const int side = lane >> 4;
  const int col  = wave * 16 + ln;

  // Preload this wave's A fragments: 14 k-steps x 2 x b128 global loads.
  Frag afr[KT];
#pragma unroll
  for (int k = 0; k < KT; ++k) {
    const unsigned* ap = Afrag + ((size_t)(k * NT + wave) * 32 + lane) * 8;
    afr[k].q[0] = *(const uint4*)(ap);
    afr[k].q[1] = *(const uint4*)(ap + 4);
  }

  // Prologue: async-prefetch u_0.
  for (int i = tid; i < 16 * 112; i += 896) {
    int b = i / 112, c = i % 112;
    async_g2l_b128(&uB[0][b][c * 4], U + ((size_t)b * SEQ) * STATE + c * 4);
  }

  int cur = 0;
  for (int t = 0; t < SEQ; ++t) {
    wait_async0();        // u_t staged; h_{t-1} stores drained (per wave)
    __syncthreads();      // all waves' async ops done; hb2[cur] ready

    // Async-prefetch u_{t+1} into the other buffer.
    if (t + 1 < SEQ) {
      for (int i = tid; i < 16 * 112; i += 896) {
        int b = i / 112, c = i % 112;
        async_g2l_b128(&uB[cur ^ 1][b][c * 4],
                       U + ((size_t)b * SEQ + (t + 1)) * STATE + c * 4);
      }
    }

    // C operand init = u_t tile.
    v8f acc;
#pragma unroll
    for (int r = 0; r < 8; ++r) acc[r] = uB[cur][r + 8 * side][col];

    if (t > 0) {
      // Software-pipelined serial WMMA chain: h-fragment k+1 loads overlap
      // WMMA k (the latency-critical path of the whole problem).
      const unsigned* hbase = &hb2[cur][ln][0];
      Frag ha[2];
      ha[0].q[0] = *(const uint4*)(hbase + 8 * side);
      ha[0].q[1] = *(const uint4*)(hbase + 8 * side + 4);
#pragma unroll
      for (int k = 0; k < KT; ++k) {
        if (k + 1 < KT) {
          const unsigned* hrow = hbase + 16 * (k + 1) + 8 * side;
          ha[(k + 1) & 1].q[0] = *(const uint4*)(hrow);
          ha[(k + 1) & 1].q[1] = *(const uint4*)(hrow + 4);
        }
        acc = __builtin_amdgcn_wmma_f32_16x16x32_bf16(
            false, ha[k & 1].b, false, afr[k].b, (short)0, acc, false, false);
      }
    }

    // h_t -> f32 LDS mirror.
#pragma unroll
    for (int r = 0; r < 8; ++r) hF[r + 8 * side][col] = acc[r];
    __syncthreads();

    // Pack h_t for step t+1 and async-store h_t over u_t in global.
    const int nxt = cur ^ 1;
    for (int i = tid; i < 16 * KP; i += 896) {
      int m = i / KP, kp = i % KP;
      hb2[nxt][m][kp] = pack2(hF[m][2 * kp], hF[m][2 * kp + 1]);
    }
    for (int i = tid; i < 16 * 112; i += 896) {
      int b = i / 112, c = i % 112;
      async_l2g_b128(U + ((size_t)b * SEQ + t) * STATE + c * 4,
                     &hF[b][c * 4]);
    }
    cur = nxt;
  }
}

// ---------------------------------------------------------------------------
// Host launcher. Workspace (~113.2 MiB):
//   117,440,512 B : U / H  (65536 x 448 f32, overwritten in place)
//   3 x 401,408 B : Afrag, Bfrag, Cfrag (fragment-major packed bf16)
// ---------------------------------------------------------------------------
extern "C" void kernel_launch(void* const* d_in, const int* in_sizes, int n_in,
                              void* d_out, int out_size, void* d_ws,
                              size_t ws_size, hipStream_t stream) {
  (void)in_sizes; (void)n_in; (void)out_size; (void)ws_size;
  const float* X = (const float*)d_in[0];
  const float* A = (const float*)d_in[1];
  const float* B = (const float*)d_in[2];
  const float* C = (const float*)d_in[3];
  float* Y = (float*)d_out;

  char* ws = (char*)d_ws;
  float* U = (float*)ws;
  size_t uBytes = (size_t)MROWS * STATE * sizeof(float);
  unsigned* Afrag = (unsigned*)(ws + uBytes);
  unsigned* Bfrag = Afrag + (size_t)KT * FRAG_TILE;
  unsigned* Cfrag = Bfrag + (size_t)KT * FRAG_TILE;

  prep_pack<<<(KT * FRAG_TILE + 255) / 256, 256, 0, stream>>>(
      A, B, C, Afrag, Bfrag, Cfrag);
  gemm_bf16_448<<<MROWS / 128, 256, 0, stream>>>(X, Bfrag, U);  // U = X @ B
  ssm_recurrence<<<1, 896, 0, stream>>>(U, Afrag);              // U -> H
  gemm_bf16_448<<<MROWS / 128, 256, 0, stream>>>(U, Cfrag, Y);  // Y = H @ C^T
}
